// Teacher_2layersGRU_img_to_word_50852412784889
// MI455X (gfx1250) — matrix operand order repaired and checked by reading, not compiled, not run on hardware
//
#include <hip/hip_runtime.h>

typedef __bf16 bf16_t;
typedef __attribute__((ext_vector_type(16))) __bf16 v16bf;
typedef __attribute__((ext_vector_type(8)))  __bf16 v8bf;
typedef __attribute__((ext_vector_type(8)))  float  v8f;

constexpr int Hd = 512;     // hidden dim
constexpr int G3 = 1536;    // 3*Hd (gate dim)
constexpr int Bn = 16;      // batch
constexpr int Tn = 38;      // total tokens
constexpr int Vn = 32000;   // vocab
constexpr int GRU_WGS = 4;  // workgroups in persistent GRU kernel (32 waves total)
constexpr int NBAR = 2048;  // barrier slots (need ~815)
constexpr int LDS_LD = 520; // padded LDS row stride (bank-staggering, 16B aligned)

__device__ __forceinline__ v8f wmma_bf16(v16bf a, v16bf b, v8f c) {
  return __builtin_amdgcn_wmma_f32_16x16x32_bf16(false, a, false, b, (short)0, c, false, false);
}

// A-fragment 16x32 bf16 from row-major [16 x ld]; contiguous k order (same as load_b_frag):
// lane L: M=L%16; K=(L/16)*16+0..15
__device__ __forceinline__ v16bf load_a_frag(const bf16_t* base, int ld, int k0) {
  const int lane = threadIdx.x & 31;
  const bf16_t* p = base + (size_t)(lane & 15) * ld + k0 + ((lane >> 4) << 4);
  v8bf lo = *(const v8bf*)p;
  v8bf hi = *(const v8bf*)(p + 8);
  return __builtin_shufflevector(lo, hi, 0,1,2,3,4,5,6,7,8,9,10,11,12,13,14,15);
}

// B-fragment 32x16 (KxN) from row-major W[n][k]; lane L: N=L%16, K=(L/16)*16+0..15
__device__ __forceinline__ v16bf load_b_frag(const bf16_t* W, int ntile, int ld, int k0) {
  const int lane = threadIdx.x & 31;
  return *(const v16bf*)(W + (size_t)(ntile * 16 + (lane & 15)) * ld + k0 + ((lane >> 4) << 4));
}

__device__ __forceinline__ float sigm(float x) { return 1.f / (1.f + __expf(-x)); }

// device-scope counting barrier (slots zeroed by init kernel each launch)
__device__ __forceinline__ void gbar(unsigned* bar, int slot) {
  __threadfence();
  __syncthreads();
  if (threadIdx.x == 0) {
    __hip_atomic_fetch_add(bar + slot, 1u, __ATOMIC_RELEASE, __HIP_MEMORY_SCOPE_AGENT);
    while (__hip_atomic_load(bar + slot, __ATOMIC_ACQUIRE, __HIP_MEMORY_SCOPE_AGENT) <
           (unsigned)GRU_WGS) {
      __builtin_amdgcn_s_sleep(1);
    }
  }
  __syncthreads();
  __threadfence();
}

__global__ void cvt_kernel(const float* __restrict__ src, bf16_t* __restrict__ dst, int n) {
  for (int i = blockIdx.x * blockDim.x + threadIdx.x; i < n; i += gridDim.x * blockDim.x)
    dst[i] = (bf16_t)src[i];
}

__global__ void init_kernel(unsigned* __restrict__ bar,
                            const float* __restrict__ feat, const float* __restrict__ embed,
                            float* __restrict__ H1f, float* __restrict__ H2f,
                            bf16_t* __restrict__ H1b, bf16_t* __restrict__ H2b,
                            bf16_t* __restrict__ Xb) {
  int i = blockIdx.x * blockDim.x + threadIdx.x;  // grid covers 8192
  if (i < NBAR) bar[i] = 0u;
  if (i < Bn * Hd) {
    float f = feat[i];
    H1f[i] = f; H2f[i] = f;
    H1b[i] = (bf16_t)f; H2b[i] = (bf16_t)f;      // parity-0 buffers
    Xb[i] = (bf16_t)embed[i & (Hd - 1)];         // X[0] = embed[<SOS>] broadcast
  }
}

// Persistent 2-layer GRU with software-pipelined layer chains:
// phase { A(t), B(t-1) } needs only ONE device barrier, because
//   A(t) <- A(t-1) h1 ; B(t-1) <- A(t-1) h1' (same buffer!) and B(t-2) h2.
__global__ __launch_bounds__(256) void gru_kernel(
    unsigned* __restrict__ bar,
    float* __restrict__ H1f, float* __restrict__ H2f,
    bf16_t* __restrict__ H1b, bf16_t* __restrict__ H2b,     // each 2 x [16x512]
    bf16_t* __restrict__ Xb, float* __restrict__ GI1,
    const bf16_t* __restrict__ Wih1, const bf16_t* __restrict__ Whh1,
    const bf16_t* __restrict__ Wih2, const bf16_t* __restrict__ Whh2,
    const float* __restrict__ bih1, const float* __restrict__ bhh1,
    const float* __restrict__ bih2, const float* __restrict__ bhh2) {
  const int tid  = threadIdx.x;
  const int cg   = ((int)blockIdx.x << 3) | (tid >> 5);  // column group 0..31
  const int lane = tid & 31;
  const int half = lane >> 4;
  const int j    = cg * 16 + (lane & 15);                // hidden column this lane owns
  int slot = 0;
  int pA = 0, pB = 0;  // double-buffer parities for h1/h2 bf16 operand buffers
  const v8f z8 = {0, 0, 0, 0, 0, 0, 0, 0};

  // cached layer-1 input gates for token idx: GI1[idx] = X[idx] @ Wih1^T + b_ih1
  auto gi1_phase = [&](int idx) {
    v8f a0 = z8, a1 = z8, a2 = z8;
    const bf16_t* X = Xb + (size_t)idx * Bn * Hd;
    for (int kc = 0; kc < 16; ++kc) {
      int k0 = kc * 32;
      v16bf af = load_a_frag(X, Hd, k0);
      a0 = wmma_bf16(af, load_b_frag(Wih1, cg,      Hd, k0), a0);
      a1 = wmma_bf16(af, load_b_frag(Wih1, cg + 32, Hd, k0), a1);
      a2 = wmma_bf16(af, load_b_frag(Wih1, cg + 64, Hd, k0), a2);
    }
    float* g = GI1 + (size_t)idx * Bn * G3;
#pragma unroll
    for (int p = 0; p < 8; ++p) {
      int b = p + (half << 3);
      g[(size_t)b * G3 + j]        = a0[p] + bih1[j];
      g[(size_t)b * G3 + 512 + j]  = a1[p] + bih1[512 + j];
      g[(size_t)b * G3 + 1024 + j] = a2[p] + bih1[1024 + j];
    }
  };

  // merged pipeline phase: optionally layer-1 step A(tA), optionally layer-2 step B
  auto step_phase = [&](bool doA, int tA, bool doB) {
    v8f aR = z8, aZ = z8, aN = z8;   // gh1 = h1 @ Whh1^T
    v8f iR = z8, iZ = z8, iN = z8;   // gi2 = h1' @ Wih2^T
    v8f hR = z8, hZ = z8, hN = z8;   // gh2 = h2 @ Whh2^T
    const bf16_t* h1r = H1b + (size_t)pA * (Bn * Hd);  // h1 (A input) == h1' (B input)
    const bf16_t* h2r = H2b + (size_t)pB * (Bn * Hd);
    for (int kc = 0; kc < 16; ++kc) {
      int k0 = kc * 32;
      v16bf a1 = load_a_frag(h1r, Hd, k0);  // shared by gh1 and gi2
      if (doA) {
        aR = wmma_bf16(a1, load_b_frag(Whh1, cg,      Hd, k0), aR);
        aZ = wmma_bf16(a1, load_b_frag(Whh1, cg + 32, Hd, k0), aZ);
        aN = wmma_bf16(a1, load_b_frag(Whh1, cg + 64, Hd, k0), aN);
      }
      if (doB) {
        v16bf a2 = load_a_frag(h2r, Hd, k0);
        iR = wmma_bf16(a1, load_b_frag(Wih2, cg,      Hd, k0), iR);
        iZ = wmma_bf16(a1, load_b_frag(Wih2, cg + 32, Hd, k0), iZ);
        iN = wmma_bf16(a1, load_b_frag(Wih2, cg + 64, Hd, k0), iN);
        hR = wmma_bf16(a2, load_b_frag(Whh2, cg,      Hd, k0), hR);
        hZ = wmma_bf16(a2, load_b_frag(Whh2, cg + 32, Hd, k0), hZ);
        hN = wmma_bf16(a2, load_b_frag(Whh2, cg + 64, Hd, k0), hN);
      }
    }
    if (doA) {
      const float* gi = GI1 + (size_t)tA * Bn * G3;
      bf16_t* h1o = H1b + (size_t)(pA ^ 1) * (Bn * Hd);
#pragma unroll
      for (int p = 0; p < 8; ++p) {
        int b = p + (half << 3);
        float r = sigm(gi[(size_t)b * G3 + j] + aR[p] + bhh1[j]);
        float z = sigm(gi[(size_t)b * G3 + 512 + j] + aZ[p] + bhh1[512 + j]);
        float n = tanhf(gi[(size_t)b * G3 + 1024 + j] + r * (aN[p] + bhh1[1024 + j]));
        float h0 = H1f[b * Hd + j];
        float hn = (1.f - z) * n + z * h0;
        H1f[b * Hd + j] = hn;
        h1o[b * Hd + j] = (bf16_t)hn;
      }
    }
    if (doB) {
      bf16_t* h2o = H2b + (size_t)(pB ^ 1) * (Bn * Hd);
#pragma unroll
      for (int p = 0; p < 8; ++p) {
        int b = p + (half << 3);
        float r = sigm(iR[p] + bih2[j] + hR[p] + bhh2[j]);
        float z = sigm(iZ[p] + bih2[512 + j] + hZ[p] + bhh2[512 + j]);
        float n = tanhf(iN[p] + bih2[1024 + j] + r * (hN[p] + bhh2[1024 + j]));
        float h0 = H2f[b * Hd + j];
        float hn = (1.f - z) * n + z * h0;
        H2f[b * Hd + j] = hn;
        h2o[b * Hd + j] = (bf16_t)hn;
      }
    }
  };

  gi1_phase(0);
  gbar(bar, slot++);

  for (int k = 0; k < Tn - 1; ++k) {
    // pipeline fill: A(0) alone
    step_phase(true, 0, false); gbar(bar, slot++); pA ^= 1;
    // steady state: { A(t), B(t-1) } per barrier
    for (int t = 1; t <= k; ++t) {
      step_phase(true, t, true); gbar(bar, slot++); pA ^= 1; pB ^= 1;
    }
    // pipeline drain: B(k) alone
    step_phase(false, 0, true); gbar(bar, slot++); pB ^= 1;

    // append X[k+1] = h2 (bf16), then cache its layer-1 input gates
    for (int e = (int)(blockIdx.x * 256) + tid; e < Bn * Hd; e += GRU_WGS * 256)
      Xb[(size_t)(k + 1) * Bn * Hd + e] = (bf16_t)H2f[e];
    gbar(bar, slot++);
    gi1_phase(k + 1);
    gbar(bar, slot++);
  }
}

// out[b, v, t] = sum_h X[t,b,h] * Wp[v,h] + bp[v]
// wave task = (vocab tile, t-group of 8); all 8 waves of a block share the same
// t-group, so the X slice is staged once in LDS (padded stride -> no bank camp).
__global__ __launch_bounds__(256) void proj_kernel(const bf16_t* __restrict__ Xb,
                                                   const bf16_t* __restrict__ Wp,
                                                   const float* __restrict__ bp,
                                                   float* __restrict__ out) {
  extern __shared__ bf16_t xs[];  // [tcnt][16][LDS_LD]
  const int w    = (int)blockIdx.x * 8 + (threadIdx.x >> 5);
  const int vt   = w % 2000;
  const int t0   = (w / 2000) * 8;                 // uniform per block
  const int tcnt = (Tn - t0) < 8 ? (Tn - t0) : 8;
  const int lane = threadIdx.x & 31;
  const int half = lane >> 4;

  // cooperative LDS fill: 16B chunks
  for (int c = threadIdx.x; c < tcnt * 16 * (Hd / 8); c += 256) {
    int ti = c / (16 * (Hd / 8));
    int r  = (c / (Hd / 8)) & 15;
    int cc = c % (Hd / 8);
    v8bf v = *(const v8bf*)(Xb + (size_t)(t0 + ti) * Bn * Hd + r * Hd + cc * 8);
    *(v8bf*)(xs + (size_t)ti * 16 * LDS_LD + r * LDS_LD + cc * 8) = v;
  }
  __syncthreads();

  v8f acc[8];
#pragma unroll
  for (int i = 0; i < 8; ++i) acc[i] = (v8f){0, 0, 0, 0, 0, 0, 0, 0};

  const int arow = (lane & 15) * LDS_LD + (half << 4);
  for (int kc = 0; kc < 16; ++kc) {
    int k0 = kc * 32;
    v16bf bfrag = load_b_frag(Wp, vt, Hd, k0);     // held across 8 time steps
#pragma unroll
    for (int ti = 0; ti < 8; ++ti) {
      if (ti < tcnt) {
        const bf16_t* p = xs + (size_t)ti * 16 * LDS_LD + arow + k0;
        v8bf lo = *(const v8bf*)p;
        v8bf hi = *(const v8bf*)(p + 8);
        v16bf af = __builtin_shufflevector(lo, hi, 0,1,2,3,4,5,6,7,8,9,10,11,12,13,14,15);
        acc[ti] = wmma_bf16(af, bfrag, acc[ti]);
      }
    }
  }
  const int v = vt * 16 + (lane & 15);
  const float bias = bp[v];
#pragma unroll
  for (int ti = 0; ti < 8; ++ti) {
    if (ti < tcnt) {
      int t = t0 + ti;
#pragma unroll
      for (int p = 0; p < 8; ++p) {
        int b = p + (half << 3);
        out[(size_t)b * Vn * Tn + (size_t)v * Tn + t] = acc[ti][p] + bias;
      }
    }
  }
}

extern "C" void kernel_launch(void* const* d_in, const int* in_sizes, int n_in,
                              void* d_out, int out_size, void* d_ws, size_t ws_size,
                              hipStream_t stream) {
  (void)in_sizes; (void)n_in; (void)out_size; (void)ws_size;
  const float* feat   = (const float*)d_in[0];
  const float* embed  = (const float*)d_in[1];
  const float* w_ih1  = (const float*)d_in[2];
  const float* w_hh1  = (const float*)d_in[3];
  const float* b_ih1  = (const float*)d_in[4];
  const float* b_hh1  = (const float*)d_in[5];
  const float* w_ih2  = (const float*)d_in[6];
  const float* w_hh2  = (const float*)d_in[7];
  const float* b_ih2  = (const float*)d_in[8];
  const float* b_hh2  = (const float*)d_in[9];
  const float* w_proj = (const float*)d_in[10];
  const float* b_proj = (const float*)d_in[11];

  char* ws = (char*)d_ws;
  size_t o = 0;
  auto take = [&](size_t bytes) -> char* {
    char* r = ws + o;
    o = (o + bytes + 255) & ~(size_t)255;
    return r;
  };
  unsigned* bar = (unsigned*)take((size_t)NBAR * 4);
  float* H1f    = (float*)take((size_t)Bn * Hd * 4);
  float* H2f    = (float*)take((size_t)Bn * Hd * 4);
  bf16_t* H1b   = (bf16_t*)take((size_t)2 * Bn * Hd * 2);
  bf16_t* H2b   = (bf16_t*)take((size_t)2 * Bn * Hd * 2);
  bf16_t* Xb    = (bf16_t*)take((size_t)Tn * Bn * Hd * 2);
  float* GI1    = (float*)take((size_t)Tn * Bn * G3 * 4);
  bf16_t* Wih1b = (bf16_t*)take((size_t)G3 * Hd * 2);
  bf16_t* Whh1b = (bf16_t*)take((size_t)G3 * Hd * 2);
  bf16_t* Wih2b = (bf16_t*)take((size_t)G3 * Hd * 2);
  bf16_t* Whh2b = (bf16_t*)take((size_t)G3 * Hd * 2);
  bf16_t* Wpb   = (bf16_t*)take((size_t)Vn * Hd * 2);

  init_kernel<<<32, 256, 0, stream>>>(bar, feat, embed, H1f, H2f, H1b, H2b, Xb);

  cvt_kernel<<<3072, 256, 0, stream>>>(w_ih1, Wih1b, G3 * Hd);
  cvt_kernel<<<3072, 256, 0, stream>>>(w_hh1, Whh1b, G3 * Hd);
  cvt_kernel<<<3072, 256, 0, stream>>>(w_ih2, Wih2b, G3 * Hd);
  cvt_kernel<<<3072, 256, 0, stream>>>(w_hh2, Whh2b, G3 * Hd);
  cvt_kernel<<<4096, 256, 0, stream>>>(w_proj, Wpb, Vn * Hd);

  gru_kernel<<<GRU_WGS, 256, 0, stream>>>(bar, H1f, H2f, H1b, H2b, Xb, GI1,
                                          Wih1b, Whh1b, Wih2b, Whh2b,
                                          b_ih1, b_hh1, b_ih2, b_hh2);

  // dynamic LDS: 8 time steps x 16 rows x padded 520 cols of bf16 = 133120 B
  proj_kernel<<<1250, 256, (size_t)8 * 16 * LDS_LD * sizeof(bf16_t), stream>>>(
      Xb, Wpb, b_proj, (float*)d_out);
}
